// STGCN_77936476553441
// MI455X (gfx1250) — matrix operand order repaired and checked by reading, not compiled, not run on hardware
//
#include <hip/hip_runtime.h>

typedef __attribute__((ext_vector_type(16))) _Float16 v16h;
typedef __attribute__((ext_vector_type(8)))  float    v8f;
typedef __attribute__((ext_vector_type(4)))  unsigned v4u;

#define EPSBN 1e-5f

static __device__ __forceinline__ float bn_inv() { return rsqrtf(1.0f + EPSBN); }

// ================= data_bn: x[B,T,V,3] (chunk, f32) -> h[3, Nc] (f16) =======
__global__ void k_databn(const float* __restrict__ x, _Float16* __restrict__ h,
                         const float* __restrict__ g, const float* __restrict__ b,
                         int Nc)
{
  int idx = blockIdx.x * blockDim.x + threadIdx.x;
  if (idx >= 3 * Nc) return;
  int ci = idx / Nc;
  int n  = idx - ci * Nc;
  int v  = n % 24;
  int ch = v * 3 + ci;
  float r = x[(size_t)n * 3 + ci] * (bn_inv() * g[ch]) + b[ch];
  h[(size_t)ci * Nc + n] = (_Float16)r;
}

// ================= graph (spatial) mixing (f16 io, f32 math) ================
// mode 0 (input side, layer0): out[(k*C+c), bt*V+w] = sum_v in[c, bt*V+v]*Ae[k,v,w]
// mode 1 (output side):        out[c, bt*V+w] = sum_{k,v} in[(k*C+c), bt*V+v]*Ae[k,v,w]
//                              then optional BN(channel c) + ReLU
__global__ void k_amix(const _Float16* __restrict__ in, _Float16* __restrict__ out,
                       const float* __restrict__ Amat, const float* __restrict__ ei,
                       int C, int Nc, int mode,
                       const float* __restrict__ bng, const float* __restrict__ bnb,
                       int relu)
{
  __shared__ float sAe[3 * 24 * 24];
  for (int i = threadIdx.x; i < 3 * 24 * 24; i += blockDim.x)
    sAe[i] = Amat[i] * ei[i];
  __syncthreads();
  const int V = 24;
  int Nbt = Nc / V;
  int gid = blockIdx.x * blockDim.x + threadIdx.x;
  if (mode == 1) {
    if (gid >= C * Nbt) return;
    int bt = gid % Nbt, c = gid / Nbt;
    float acc[24];
#pragma unroll
    for (int w = 0; w < 24; ++w) acc[w] = 0.0f;
    for (int k = 0; k < 3; ++k) {
      const _Float16* row = in + (size_t)(k * C + c) * Nc + bt * V;
      for (int v = 0; v < 24; ++v) {
        float xv = (float)row[v];
        const float* Ae = &sAe[(k * 24 + v) * 24];
#pragma unroll
        for (int w = 0; w < 24; ++w) acc[w] = fmaf(xv, Ae[w], acc[w]);
      }
    }
    float s = 1.0f, sh = 0.0f;
    if (bng) { s = bn_inv() * bng[c]; sh = bnb[c]; }
    _Float16* orow = out + (size_t)c * Nc + bt * V;
#pragma unroll
    for (int w = 0; w < 24; ++w) {
      float vv = acc[w] * s + sh;
      if (relu) vv = fmaxf(vv, 0.0f);
      orow[w] = (_Float16)vv;
    }
  } else {
    if (gid >= 3 * C * Nbt) return;
    int bt = gid % Nbt, r = gid / Nbt;
    int c = r % C, k = r / C;
    float acc[24];
#pragma unroll
    for (int w = 0; w < 24; ++w) acc[w] = 0.0f;
    const _Float16* row = in + (size_t)c * Nc + bt * V;
    for (int v = 0; v < 24; ++v) {
      float xv = (float)row[v];
      const float* Ae = &sAe[(k * 24 + v) * 24];
#pragma unroll
      for (int w = 0; w < 24; ++w) acc[w] = fmaf(xv, Ae[w], acc[w]);
    }
    _Float16* orow = out + (size_t)(k * C + c) * Nc + bt * V;
#pragma unroll
    for (int w = 0; w < 24; ++w) orow[w] = (_Float16)acc[w];
  }
}

// ====== weight pack: f32 [M][Ksrc] -> f16 [M][Kp], zero pad Ksrc..Kp ========
__global__ void k_packw(const float* __restrict__ src, _Float16* __restrict__ dst,
                        int M, int Ksrc, int Kp)
{
  int idx = blockIdx.x * blockDim.x + threadIdx.x;
  if (idx >= M * Kp) return;
  int m = idx / Kp, k = idx - m * Kp;
  dst[idx] = (k < Ksrc) ? (_Float16)src[(size_t)m * Ksrc + k] : (_Float16)0.0f;
}

// ====== layer0 repack+pack: W0r[c, k*3+ci] = gcn_w[(k*256+c)*3+ci], Kp=32 ===
__global__ void k_packw0(const float* __restrict__ w, _Float16* __restrict__ dst)
{
  int idx = blockIdx.x * blockDim.x + threadIdx.x;
  if (idx >= 256 * 32) return;
  int c = idx >> 5, kk = idx & 31;
  float v = 0.0f;
  if (kk < 9) {
    int k = kk / 3, ci = kk % 3;
    v = w[(k * 256 + c) * 3 + ci];
  }
  dst[idx] = (_Float16)v;
}

// ====== layer0 node bias: bW[c,w] = sum_k b[k*256+c]*sum_v (A*ei)[k,v,w] ====
__global__ void k_biasw0(const float* __restrict__ gb, const float* __restrict__ Amat,
                         const float* __restrict__ ei, float* __restrict__ bW)
{
  int idx = blockIdx.x * blockDim.x + threadIdx.x;
  if (idx >= 256 * 24) return;
  int w = idx % 24, c = idx / 24;
  float acc = 0.0f;
  for (int k = 0; k < 3; ++k) {
    float s = 0.0f;
    for (int v = 0; v < 24; ++v) {
      int a = (k * 24 + v) * 24 + w;
      s += Amat[a] * ei[a];
    }
    acc += gb[k * 256 + c] * s;
  }
  bW[c * 24 + w] = acc;
}

// ================= WMMA GEMM, f16 in / f16 out, implicit temporal im2col ====
// Y[M,Nc] = epilogue( A[M, Kp] x Xim2col[Kbase*taps, Nc] ), Kp = pad32(Kbase*taps)
// B column k = ci*taps+dt reads X[ci, n + (dt - taps/2)*V], zero outside [0,T).
// Epilogue order: +bias[m], +biasW[m, n%24], BN(m), +resid[m,n], ReLU.
// A tiles: double-buffered LDS, staged by GLOBAL_LOAD_ASYNC_TO_LDS_B128 (ASYNCcnt),
// overlapped with the WMMAs of the previous k-step. B tiles: reg->ds (transpose+im2col).
__launch_bounds__(128)
__global__ void k_gemm_wmma(const _Float16* __restrict__ A,
                            const _Float16* __restrict__ X,
                            _Float16* __restrict__ Y,
                            int M, int Kbase, int taps, int Nc, int T,
                            const float* __restrict__ bias,
                            const float* __restrict__ biasW,
                            const float* __restrict__ bng,
                            const float* __restrict__ bnb,
                            const _Float16* __restrict__ resid,
                            int relu)
{
  (void)M;
  const int V = 24;
  __shared__ unsigned sAd[2][64 * 20];   // [buf][m][kpair 0..15, pad 4] (80B rows)
  __shared__ _Float16 sB[64][40];        // [n][k 0..31, pad 8] (80B rows) == X-tile^T
  const int tid  = threadIdx.x;
  const int lane = tid & 31;
  const int wave = tid >> 5;
  const int mt = blockIdx.y * 64;
  const int nt = blockIdx.x * 64;
  const int Ktot = Kbase * taps;
  const int Kp   = (Ktot + 31) & ~31;    // weight pack stride (pre-padded)
  const int htaps = taps >> 1;

  v8f acc[4];
#pragma unroll
  for (int s = 0; s < 4; ++s)
#pragma unroll
    for (int g = 0; g < 8; ++g) acc[s][g] = 0.0f;

  _Float16 bReg[16];

  // ---- A tile: async DMA global -> LDS (2 x b128 per thread, 16B aligned) ----
  auto asyncFetchA = [&](int buf, int k0) {
#pragma unroll
    for (int i = 0; i < 2; ++i) {
      int idx = tid + i * 128;           // 256 x b128: 64 rows x 4 quads
      int m = idx >> 2, q = idx & 3;
      unsigned dst = (unsigned)(size_t)&sAd[buf][m * 20 + q * 4];
      unsigned long long src =
          (unsigned long long)(size_t)(A + (size_t)(mt + m) * Kp + k0 + q * 8);
      asm volatile("global_load_async_to_lds_b128 %0, %1, off"
                   :: "v"(dst), "v"(src) : "memory");
    }
  };
  // ---- B tile: gather (with im2col shift + bounds select) into regs ----
  auto fetchB = [&](int k0) {
#pragma unroll
    for (int i = 0; i < 16; ++i) {
      int idx = tid + i * 128;           // 2048 halves: 32 k x 64 n
      int kk = idx >> 6, nn = idx & 63;
      int kg = k0 + kk;
      bool ok = kg < Ktot;
      size_t ix = 0;
      if (taps == 1) {
        ix = ok ? ((size_t)kg * Nc + nt + nn) : (size_t)0;
      } else {
        int ci = kg / taps;
        int dt = kg - ci * taps;
        int shift = dt - htaps;
        int n = nt + nn;
        int t = (n / V) % T;
        int ts = t + shift;
        ok = ok && (ts >= 0) && (ts < T);
        ix = ok ? ((size_t)ci * Nc + n + shift * V) : (size_t)0;
      }
      _Float16 ld = X[ix];
      bReg[i] = ok ? ld : (_Float16)0.0f;
    }
  };
  auto stageB = [&]() {
#pragma unroll
    for (int i = 0; i < 16; ++i) {
      int idx = tid + i * 128;
      int kk = idx >> 6, nn = idx & 63;
      sB[nn][kk] = bReg[i];              // transpose into fragment-order rows
    }
  };

  asyncFetchA(0, 0);
  fetchB(0);

  for (int k0 = 0; k0 < Kp; k0 += 32) {
    int cur = (k0 >> 5) & 1;
    stageB();
    asm volatile("s_wait_asynccnt 0x0" ::: "memory");   // own async A done
    __syncthreads();                                    // all waves' tiles visible

    if (k0 + 32 < Kp) {
      asyncFetchA(cur ^ 1, k0 + 32);    // DMA next A tile under this step's math
      fetchB(k0 + 32);
    }

    // A fragment: two b128 (ISA 16x32 f16 layout)
    union { v16h h; v4u q[2]; } af;
    {
      const _Float16* sAh = (const _Float16*)sAd[cur];
      int ml = wave * 16 + (lane & 15);
      int grp = lane >> 4;
      const _Float16* p = sAh + ml * 40 + grp * 8;
      af.q[0] = *reinterpret_cast<const v4u*>(p);
      af.q[1] = *reinterpret_cast<const v4u*>(p + 16);
    }
    // 4 N-subtiles: two b128 each (ISA 32x16 f16 layout)
#pragma unroll
    for (int s = 0; s < 4; ++s) {
      union { v16h h; v4u q[2]; } bf;
      int nl = s * 16 + (lane & 15);
      int kh = lane >> 4;
      const _Float16* p = &sB[nl][kh * 8];
      bf.q[0] = *reinterpret_cast<const v4u*>(p);
      bf.q[1] = *reinterpret_cast<const v4u*>(p + 16);
      acc[s] = __builtin_amdgcn_wmma_f32_16x16x32_f16(
          false, af.h, false, bf.h, (short)0, acc[s], false, false);
    }
    __syncthreads();
  }

  // epilogue + store (C/D layout: vgpr g -> m = g + 8*(lane/16), n = lane%16)
  const float inv = bn_inv();
  int grp = lane >> 4;
  int nlb = lane & 15;
#pragma unroll
  for (int s = 0; s < 4; ++s) {
    int n = nt + s * 16 + nlb;
#pragma unroll
    for (int g = 0; g < 8; ++g) {
      int m = mt + wave * 16 + g + 8 * grp;
      float v = acc[s][g];
      if (bias)  v += bias[m];
      if (biasW) v += biasW[m * V + (n % V)];
      if (bng)   v = v * (inv * bng[m]) + bnb[m];
      if (resid) v += (float)resid[(size_t)m * Nc + n];
      if (relu)  v = fmaxf(v, 0.0f);
      Y[(size_t)m * Nc + n] = (_Float16)v;
    }
  }
}

// ============ final FC 64->6 (f16 in, f32 out), transposed store [n,6] ======
__global__ void k_fcn(const _Float16* __restrict__ h, const float* __restrict__ w,
                      const float* __restrict__ b, float* __restrict__ out, int Nc)
{
  __shared__ float sw[6 * 64];
  __shared__ float sb[6];
  for (int i = threadIdx.x; i < 384; i += blockDim.x) sw[i] = w[i];
  if (threadIdx.x < 6) sb[threadIdx.x] = b[threadIdx.x];
  __syncthreads();
  int n = blockIdx.x * blockDim.x + threadIdx.x;
  if (n >= Nc) return;
  float acc[6];
#pragma unroll
  for (int o = 0; o < 6; ++o) acc[o] = sb[o];
  for (int c = 0; c < 64; ++c) {
    float xv = (float)h[(size_t)c * Nc + n];
#pragma unroll
    for (int o = 0; o < 6; ++o) acc[o] = fmaf(xv, sw[o * 64 + c], acc[o]);
  }
#pragma unroll
  for (int o = 0; o < 6; ++o) out[(size_t)n * 6 + o] = acc[o];
}

static inline int cdiv(int a, int b) { return (a + b - 1) / b; }

extern "C" void kernel_launch(void* const* d_in, const int* in_sizes, int n_in,
                              void* d_out, int out_size, void* d_ws, size_t ws_size,
                              hipStream_t stream)
{
  (void)in_sizes; (void)n_in; (void)out_size;
  const float* x        = (const float*)d_in[0];
  const float* Amat     = (const float*)d_in[1];
  const float* dbn_g    = (const float*)d_in[2];
  const float* dbn_b    = (const float*)d_in[3];
  const float* ei0      = (const float*)d_in[4];
  const float* ei1      = (const float*)d_in[5];
  const float* ei2      = (const float*)d_in[6];
  const float* l0_gcn_w = (const float*)d_in[7];
  const float* l0_gcn_b = (const float*)d_in[8];
  const float* l0_bn1_g = (const float*)d_in[9];
  const float* l0_bn1_b = (const float*)d_in[10];
  const float* l0_tcn_w = (const float*)d_in[11];
  const float* l0_tcn_b = (const float*)d_in[12];
  const float* l0_bn2_g = (const float*)d_in[13];
  const float* l0_bn2_b = (const float*)d_in[14];
  const float* l1_gcn_w = (const float*)d_in[15];
  const float* l1_gcn_b = (const float*)d_in[16];
  const float* l1_bn1_g = (const float*)d_in[17];
  const float* l1_bn1_b = (const float*)d_in[18];
  const float* l1_tcn_w = (const float*)d_in[19];
  const float* l1_tcn_b = (const float*)d_in[20];
  const float* l1_bn2_g = (const float*)d_in[21];
  const float* l1_bn2_b = (const float*)d_in[22];
  const float* l1_res_w = (const float*)d_in[23];
  const float* l1_res_b = (const float*)d_in[24];
  const float* l1_res_bn_g = (const float*)d_in[25];
  const float* l1_res_bn_b = (const float*)d_in[26];
  const float* l2_gcn_w = (const float*)d_in[27];
  const float* l2_gcn_b = (const float*)d_in[28];
  const float* l2_bn1_g = (const float*)d_in[29];
  const float* l2_bn1_b = (const float*)d_in[30];
  const float* l2_tcn_w = (const float*)d_in[31];
  const float* l2_tcn_b = (const float*)d_in[32];
  const float* l2_bn2_g = (const float*)d_in[33];
  const float* l2_bn2_b = (const float*)d_in[34];
  const float* l2_res_w = (const float*)d_in[35];
  const float* l2_res_b = (const float*)d_in[36];
  const float* l2_res_bn_g = (const float*)d_in[37];
  const float* l2_res_bn_b = (const float*)d_in[38];
  const float* fcn_w    = (const float*)d_in[39];
  const float* fcn_b    = (const float*)d_in[40];
  float* out = (float*)d_out;
  char* base = (char*)d_ws;

  const int Bn = 32, T = 256, V = 24;

  // ---- workspace layout (16B-aligned sub-buffers) ----
  size_t off = 0;
  auto allocH = [&](size_t halves) -> _Float16* {
    off = (off + 15) & ~(size_t)15;
    _Float16* p = (_Float16*)(base + off);
    off += halves * sizeof(_Float16);
    return p;
  };
  auto allocF = [&](size_t floats) -> float* {
    off = (off + 15) & ~(size_t)15;
    float* p = (float*)(base + off);
    off += floats * sizeof(float);
    return p;
  };

  // packed f16 weights (K padded to 32)
  _Float16* pW0r  = allocH(256 * 32);
  _Float16* pTcn0 = allocH((size_t)256 * 1280);
  _Float16* pGcn1 = allocH((size_t)384 * 256);
  _Float16* pRes1 = allocH((size_t)128 * 256);
  _Float16* pTcn1 = allocH((size_t)128 * 640);
  _Float16* pGcn2 = allocH((size_t)192 * 128);
  _Float16* pRes2 = allocH((size_t)64 * 128);
  _Float16* pTcn2 = allocH((size_t)64 * 320);
  float*    bW0   = allocF(256 * 24);
  size_t fixed = (off + 15) & ~(size_t)15;

  // pick largest whole-batch chunk that fits (1676 f16 activation rows)
  int CB = 1;
  {
    const int cand[4] = {8, 4, 2, 1};
    for (int i = 0; i < 4; ++i) {
      size_t need = fixed + ((size_t)1676 * cand[i] * T * V + 64) * sizeof(_Float16);
      if (need <= ws_size) { CB = cand[i]; break; }
    }
  }
  const int Nc  = CB * T * V;          // columns per chunk (batch-contiguous)
  const int Nbt = Nc / V;
  const int nchunks = Bn / CB;
  const int NT = Nc / 64;

  _Float16* h0  = allocH((size_t)3   * Nc);
  _Float16* Xa0 = allocH((size_t)9   * Nc);
  _Float16* z0  = allocH((size_t)256 * Nc);
  _Float16* h1  = allocH((size_t)256 * Nc);
  _Float16* y1  = allocH((size_t)384 * Nc);
  _Float16* z1  = allocH((size_t)128 * Nc);
  _Float16* r1  = allocH((size_t)128 * Nc);
  _Float16* h2  = allocH((size_t)128 * Nc);
  _Float16* y2  = allocH((size_t)192 * Nc);
  _Float16* z2  = allocH((size_t)64  * Nc);
  _Float16* r2  = allocH((size_t)64  * Nc);
  _Float16* h3  = allocH((size_t)64  * Nc);

  // ---- one-time weight packing ----
  k_packw0<<<cdiv(256 * 32, 256), 256, 0, stream>>>(l0_gcn_w, pW0r);
  k_packw<<<cdiv(256 * 1280, 256), 256, 0, stream>>>(l0_tcn_w, pTcn0, 256, 1280, 1280);
  k_packw<<<cdiv(384 * 256, 256), 256, 0, stream>>>(l1_gcn_w, pGcn1, 384, 256, 256);
  k_packw<<<cdiv(128 * 256, 256), 256, 0, stream>>>(l1_res_w, pRes1, 128, 256, 256);
  k_packw<<<cdiv(128 * 640, 256), 256, 0, stream>>>(l1_tcn_w, pTcn1, 128, 640, 640);
  k_packw<<<cdiv(192 * 128, 256), 256, 0, stream>>>(l2_gcn_w, pGcn2, 192, 128, 128);
  k_packw<<<cdiv(64 * 128, 256), 256, 0, stream>>>(l2_res_w, pRes2, 64, 128, 128);
  k_packw<<<cdiv(64 * 320, 256), 256, 0, stream>>>(l2_tcn_w, pTcn2, 64, 320, 320);
  k_biasw0<<<cdiv(256 * 24, 128), 128, 0, stream>>>(l0_gcn_b, Amat, ei0, bW0);

  for (int ck = 0; ck < nchunks; ++ck) {
    const float* xc = x + (size_t)ck * Nc * 3;
    float* oc = out + (size_t)ck * Nc * 6;

    k_databn<<<cdiv(3 * Nc, 256), 256, 0, stream>>>(xc, h0, dbn_g, dbn_b, Nc);

    // ---- layer 0 (graph-mix on input side; Cin=3 -> 9 mixed channels) ----
    k_amix<<<cdiv(9 * Nbt, 128), 128, 0, stream>>>(h0, Xa0, Amat, ei0, 3, Nc, 0,
                                                   nullptr, nullptr, 0);
    k_gemm_wmma<<<dim3(NT, 4), 128, 0, stream>>>(pW0r, Xa0, z0, 256, 9, 1, Nc, T,
        nullptr, bW0, l0_bn1_g, l0_bn1_b, nullptr, 1);
    k_gemm_wmma<<<dim3(NT, 4), 128, 0, stream>>>(pTcn0, z0, h1, 256, 256, 5, Nc, T,
        l0_tcn_b, nullptr, l0_bn2_g, l0_bn2_b, nullptr, 1);

    // ---- layer 1 (256 -> 128, residual 1x1+BN) ----
    k_gemm_wmma<<<dim3(NT, 6), 128, 0, stream>>>(pGcn1, h1, y1, 384, 256, 1, Nc, T,
        l1_gcn_b, nullptr, nullptr, nullptr, nullptr, 0);
    k_amix<<<cdiv(128 * Nbt, 128), 128, 0, stream>>>(y1, z1, Amat, ei1, 128, Nc, 1,
                                                     l1_bn1_g, l1_bn1_b, 1);
    k_gemm_wmma<<<dim3(NT, 2), 128, 0, stream>>>(pRes1, h1, r1, 128, 256, 1, Nc, T,
        l1_res_b, nullptr, l1_res_bn_g, l1_res_bn_b, nullptr, 0);
    k_gemm_wmma<<<dim3(NT, 2), 128, 0, stream>>>(pTcn1, z1, h2, 128, 128, 5, Nc, T,
        l1_tcn_b, nullptr, l1_bn2_g, l1_bn2_b, r1, 1);

    // ---- layer 2 (128 -> 64, residual 1x1+BN) ----
    k_gemm_wmma<<<dim3(NT, 3), 128, 0, stream>>>(pGcn2, h2, y2, 192, 128, 1, Nc, T,
        l2_gcn_b, nullptr, nullptr, nullptr, nullptr, 0);
    k_amix<<<cdiv(64 * Nbt, 128), 128, 0, stream>>>(y2, z2, Amat, ei2, 64, Nc, 1,
                                                    l2_bn1_g, l2_bn1_b, 1);
    k_gemm_wmma<<<dim3(NT, 1), 128, 0, stream>>>(pRes2, h2, r2, 64, 128, 1, Nc, T,
        l2_res_b, nullptr, l2_res_bn_g, l2_res_bn_b, nullptr, 0);
    k_gemm_wmma<<<dim3(NT, 1), 128, 0, stream>>>(pTcn2, z2, h3, 64, 64, 5, Nc, T,
        l2_tcn_b, nullptr, l2_bn2_g, l2_bn2_b, r2, 1);

    // ---- final FC + transpose to [B,T,V,6] ----
    k_fcn<<<cdiv(Nc, 128), 128, 0, stream>>>(h3, fcn_w, fcn_b, oc, Nc);
  }
}
